// Malconv_88880053223578
// MI455X (gfx1250) — matrix operand order repaired and checked by reading, not compiled, not run on hardware
//
#include <hip/hip_runtime.h>

// ---------------- problem constants ----------------
#define BATCH   64
#define MAX_LEN 200000
#define WIN     500
#define NWIN    400        // MAX_LEN / WIN
#define EMB     8
#define CH      128
#define K_TOT   4000       // WIN * EMB
#define K_PAD   4032       // padded to 63 * 64 (zero-filled B => no-op contributions)
#define KCHUNKS 63         // K_PAD / 64
#define M_TOT   25600      // BATCH * NWIN

// ---------------- WMMA types ----------------
typedef __attribute__((ext_vector_type(16))) __bf16 v16bf;
typedef __attribute__((ext_vector_type(8)))  float  v8f;

union FragBF {
    v16bf v;
    uint4 q[2];
};

#define WMMA_BF16(A, B, C) \
    __builtin_amdgcn_wmma_f32_16x16x32_bf16(false, (A), false, (B), (short)0, (C), false, false)

// round-to-nearest-even f32 -> bf16 (as ushort)
__device__ __forceinline__ unsigned short f2bf(float f) {
    union { float f; unsigned u; } v;
    v.f = f;
    unsigned r = v.u + 0x7FFFu + ((v.u >> 16) & 1u);
    return (unsigned short)(r >> 16);
}

// ---------------- workspace layout ----------------
// w1t   : CH*K_PAD ushort (bf16 [CH][K_PAD], k = kk*8 + c, zero pad k>=4000)
// w2t   : CH*K_PAD ushort
// embbf : 2048 ushort     (bf16 [256][8], 16B rows)
// pooled: 8192 float      ([BATCH][CH])
#define N_WTP  (CH * K_PAD)        // 516096
#define N_EMBT (256 * EMB)         // 2048
#define N_POOL (BATCH * CH)        // 8192
#define PREP_TOTAL (2 * N_WTP + N_EMBT + N_POOL)   // 1,042,432 = 4072 * 256

// ======================================================================
// Kernel 1: pack weights to bf16 [n][k] (zero-padded), bf16 emb, zero pooled
// ======================================================================
__global__ __launch_bounds__(256) void malconv_prep(
    const float* __restrict__ w1, const float* __restrict__ w2,
    const float* __restrict__ emb,
    unsigned short* __restrict__ w1t, unsigned short* __restrict__ w2t,
    unsigned short* __restrict__ embbf, float* __restrict__ pooled)
{
    int i = blockIdx.x * 256 + threadIdx.x;
    if (i < N_WTP) {
        int o = i / K_PAD;
        int k = i - o * K_PAD;
        unsigned short v = 0;
        if (k < K_TOT) {
            int kk = k >> 3;       // position in window
            int c  = k & 7;        // embedding channel
            v = f2bf(w1[o * K_TOT + c * WIN + kk]);
        }
        w1t[i] = v;
    } else if (i < 2 * N_WTP) {
        int j = i - N_WTP;
        int o = j / K_PAD;
        int k = j - o * K_PAD;
        unsigned short v = 0;
        if (k < K_TOT) {
            int kk = k >> 3;
            int c  = k & 7;
            v = f2bf(w2[o * K_TOT + c * WIN + kk]);
        }
        w2t[j] = v;
    } else if (i < 2 * N_WTP + N_EMBT) {
        int j = i - 2 * N_WTP;
        embbf[j] = f2bf(emb[j]);
    } else if (i < PREP_TOTAL) {
        pooled[i - 2 * N_WTP - N_EMBT] = 0.0f;
    }
}

// ======================================================================
// Kernel 2: fused embed-gather + dual GEMM (WMMA bf16) + gate + max-pool
//   grid = 800 blocks (32 windows each), block = 256 threads (8 waves)
//   wave w: m-subtile = w&1 (16 rows), n-pair = w>>1 (32 of 128 channels)
//   K loop: 63 chunks of 64; ping-pong LDS (1 barrier/chunk); all 10
//   fragments of a chunk loaded into distinct registers before 8 WMMAs
// ======================================================================
__global__ __launch_bounds__(256) void malconv_gemm(
    const int* __restrict__ x,
    const float* __restrict__ b1, const float* __restrict__ b2,
    const unsigned short* __restrict__ w1t,
    const unsigned short* __restrict__ w2t,
    const unsigned short* __restrict__ embbf,
    float* __restrict__ pooled)
{
    __shared__ __align__(16) unsigned short Atile[2][32][64]; // ping-pong, 2x4KB

    const int tid   = threadIdx.x;
    const int m0    = blockIdx.x * 32;
    const int lane  = tid & 31;
    const int wave  = tid >> 5;
    const int msub  = wave & 1;      // which 16-row subtile
    const int npair = wave >> 1;     // which 32-channel slice
    const int half  = lane >> 4;
    const int l16   = lane & 15;

    // A-fragment row (16-bit A layout: lane<16 holds K 0-7,16-23;
    // lane>=16 holds K 8-15,24-31 of the same row)
    const int arow = msub * 16 + l16;

    // B columns for this lane (lanes 0-15: K 0-7,16-23; lanes 16-31: K 8-15,24-31)
    const int nA = npair * 32 + l16;
    const int nB = nA + 16;

    // staging: all 256 threads, each fetches one (row, token) -> 8 bf16
    const int srow = tid >> 3;       // 0..31
    const int skk  = tid & 7;        // 0..7 token within 64-k chunk
    int sm = m0 + srow;
    int sb = sm / NWIN;
    int st = sm - sb * NWIN;
    const int* xrow = x + sb * MAX_LEN + st * WIN;

    v8f acc1a = {}, acc1b = {}, acc2a = {}, acc2b = {};

    // prologue: stage chunk 0 into buffer 0
    {
        int tok = xrow[skk];
        *(uint4*)(&Atile[0][srow][skk * 8]) = *(const uint4*)(embbf + tok * EMB);
    }
    __syncthreads();

    for (int chunk = 0; chunk < KCHUNKS; ++chunk) {
        const int buf = chunk & 1;

        // stage next chunk into the other buffer (no conflict with reads)
        if (chunk + 1 < KCHUNKS) {
            int kk  = (chunk + 1) * 8 + skk;
            int kkc = kk < WIN ? kk : (WIN - 1);   // clamp; pad-k has zero B
            int tok = xrow[kkc];
            *(uint4*)(&Atile[buf ^ 1][srow][skk * 8]) =
                *(const uint4*)(embbf + tok * EMB);
        }

        // ---- hoist ALL fragment loads of this chunk into live registers ----
        const int kbase = chunk * 64;
        const unsigned short* q1a = w1t + nA * K_PAD + kbase + half * 8;
        const unsigned short* q1b = w1t + nB * K_PAD + kbase + half * 8;
        const unsigned short* q2a = w2t + nA * K_PAD + kbase + half * 8;
        const unsigned short* q2b = w2t + nB * K_PAD + kbase + half * 8;

        FragBF fa0, fa1;
        fa0.q[0] = *(const uint4*)(&Atile[buf][arow][half * 8]);
        fa0.q[1] = *(const uint4*)(&Atile[buf][arow][16 + half * 8]);
        fa1.q[0] = *(const uint4*)(&Atile[buf][arow][32 + half * 8]);
        fa1.q[1] = *(const uint4*)(&Atile[buf][arow][48 + half * 8]);

        FragBF g1a0, g2a0, g1b0, g2b0, g1a1, g2a1, g1b1, g2b1;
        g1a0.q[0] = *(const uint4*)(q1a);      g1a0.q[1] = *(const uint4*)(q1a + 16);
        g2a0.q[0] = *(const uint4*)(q2a);      g2a0.q[1] = *(const uint4*)(q2a + 16);
        g1b0.q[0] = *(const uint4*)(q1b);      g1b0.q[1] = *(const uint4*)(q1b + 16);
        g2b0.q[0] = *(const uint4*)(q2b);      g2b0.q[1] = *(const uint4*)(q2b + 16);
        g1a1.q[0] = *(const uint4*)(q1a + 32); g1a1.q[1] = *(const uint4*)(q1a + 48);
        g2a1.q[0] = *(const uint4*)(q2a + 32); g2a1.q[1] = *(const uint4*)(q2a + 48);
        g1b1.q[0] = *(const uint4*)(q1b + 32); g1b1.q[1] = *(const uint4*)(q1b + 48);
        g2b1.q[0] = *(const uint4*)(q2b + 32); g2b1.q[1] = *(const uint4*)(q2b + 48);

        // ---- 8 WMMAs; loads above can drain under partial waits ----
        acc1a = WMMA_BF16(fa0.v, g1a0.v, acc1a);
        acc2a = WMMA_BF16(fa0.v, g2a0.v, acc2a);
        acc1b = WMMA_BF16(fa0.v, g1b0.v, acc1b);
        acc2b = WMMA_BF16(fa0.v, g2b0.v, acc2b);
        acc1a = WMMA_BF16(fa1.v, g1a1.v, acc1a);
        acc2a = WMMA_BF16(fa1.v, g2a1.v, acc2a);
        acc1b = WMMA_BF16(fa1.v, g1b1.v, acc1b);
        acc2b = WMMA_BF16(fa1.v, g2b1.v, acc2b);

        __syncthreads();   // next-chunk staging done; reads of buf done
    }

    // ---- epilogue: bias, gate = relu(c1 * sigmoid(c2)), max over 8 rows ----
    const int mt = m0 + msub * 16;      // 16-row tile never straddles a batch
    const int b  = mt / NWIN;

    float bias1A = b1[nA], bias2A = b2[nA];
    float bias1B = b1[nB], bias2B = b2[nB];

    float gmaxA = 0.0f, gmaxB = 0.0f;
    #pragma unroll
    for (int i = 0; i < 8; ++i) {
        float c1 = acc1a[i] + bias1A;
        float c2 = acc2a[i] + bias2A;
        float g  = fmaxf(c1 / (1.0f + __expf(-c2)), 0.0f);
        gmaxA = fmaxf(gmaxA, g);

        c1 = acc1b[i] + bias1B;
        c2 = acc2b[i] + bias2B;
        g  = fmaxf(c1 / (1.0f + __expf(-c2)), 0.0f);
        gmaxB = fmaxf(gmaxB, g);
    }
    // gated >= 0, pooled initialized to 0 -> int-max == float-max
    atomicMax((int*)(pooled + b * CH + nA), __float_as_int(gmaxA));
    atomicMax((int*)(pooled + b * CH + nB), __float_as_int(gmaxB));
}

// ======================================================================
// Kernel 3: tiny MLP head: out = sigmoid(relu(pooled@fw1^T+fb1)@fw2^T+fb2)
// ======================================================================
__global__ void malconv_head(
    const float* __restrict__ pooled,
    const float* __restrict__ fw1, const float* __restrict__ fb1,
    const float* __restrict__ fw2, const float* __restrict__ fb2,
    float* __restrict__ out)
{
    int b = threadIdx.x;    // 64 threads
    if (b >= BATCH) return;
    const float* pr = pooled + b * CH;
    float o = fb2[0];
    for (int j = 0; j < 64; ++j) {
        const float* wr = fw1 + j * CH;
        float s = fb1[j];
        #pragma unroll 8
        for (int n = 0; n < CH; ++n) s += pr[n] * wr[n];
        s = fmaxf(s, 0.0f);
        o += s * fw2[j];
    }
    out[b] = 1.0f / (1.0f + __expf(-o));
}

// ======================================================================
extern "C" void kernel_launch(void* const* d_in, const int* in_sizes, int n_in,
                              void* d_out, int out_size, void* d_ws, size_t ws_size,
                              hipStream_t stream) {
    const int*   x   = (const int*)  d_in[0];
    const float* emb = (const float*)d_in[1];
    const float* w1  = (const float*)d_in[2];
    const float* b1  = (const float*)d_in[3];
    const float* w2  = (const float*)d_in[4];
    const float* b2  = (const float*)d_in[5];
    const float* fw1 = (const float*)d_in[6];
    const float* fb1 = (const float*)d_in[7];
    const float* fw2 = (const float*)d_in[8];
    const float* fb2 = (const float*)d_in[9];
    float* out = (float*)d_out;

    unsigned short* w1t    = (unsigned short*)d_ws;
    unsigned short* w2t    = w1t + N_WTP;
    unsigned short* embbf  = w2t + N_WTP;
    float*          pooled = (float*)(embbf + N_EMBT);

    // 1) pack weights/emb to bf16, zero pooled (every call; harness poisons ws)
    malconv_prep<<<PREP_TOTAL / 256, 256, 0, stream>>>(
        w1, w2, emb, w1t, w2t, embbf, pooled);

    // 2) fused gather + dual WMMA GEMM + gate + global max-pool
    malconv_gemm<<<M_TOT / 32, 256, 0, stream>>>(
        x, b1, b2, w1t, w2t, embbf, pooled);

    // 3) MLP head
    malconv_head<<<1, 64, 0, stream>>>(pooled, fw1, fb1, fw2, fb2, out);
}
